// GNN_50972671869116
// MI455X (gfx1250) — compile-verified, matrix operand order checked by reading, not run on hardware
//
#include <hip/hip_runtime.h>
#include <hip/hip_bf16.h>
#include <hip/hip_fp16.h>

// PointGNN (3 layers, D=64) for gfx1250: all MLPs via v_wmma_f32_16x16x32_f16,
// segment_max fused into the edge kernel via order-preserving u32 atomic max.
// Edge kernel: 10 tiles per wave with all weight B-fragments VGPR-resident.

#define N_NODES 50000
#define N_EDGES 800000
#define DD 64
#define NEG_BIG (-1e30f)
#define EDGE_TILES_PER_WAVE 10

typedef __attribute__((ext_vector_type(16))) _Float16 v16h;
typedef __attribute__((ext_vector_type(8)))  float    v8f;
typedef __attribute__((ext_vector_type(4)))  float    v4f;

union AFrag { v16h v; v4f f[2]; };

// CDNA5 16-bit A-matrix layout: lane (g = lane>>4, r = lane&15) holds row M=r.
// Half h (0..15) maps to K = h + 8*g + (h>=8 ? 8 : 0), i.e. two contiguous
// 8-half (16B) segments at K = 8g and K = 16+8g within each 32-wide K chunk.
__device__ __forceinline__ v16h load_a16(const _Float16* row, int c, int g) {
  AFrag u;
  u.f[0] = *(const v4f*)(row + 32 * c + 8 * g);
  u.f[1] = *(const v4f*)(row + 32 * c + 16 + 8 * g);
  return u.v;
}
// Pre-packed B fragment: 16 contiguous halves per lane.
__device__ __forceinline__ v16h load_b16(const _Float16* p) {
  AFrag u;
  u.f[0] = *(const v4f*)(p);
  u.f[1] = *(const v4f*)(p + 8);
  return u.v;
}

// Wave-local LDS ordering: lanes of a wave share one PC, so only DS-counter
// completion is needed before cross-lane LDS reads (CDNA5 split counters).
__device__ __forceinline__ void wave_lds_sync() {
  asm volatile("s_wait_dscnt 0x0" ::: "memory");
}

// Order-preserving float<->u32 so atomicMax(u32) implements float max.
__device__ __forceinline__ unsigned encf(float f) {
  unsigned b = __float_as_uint(f);
  return (b >> 31) ? ~b : (b | 0x80000000u);
}
__device__ __forceinline__ float decf(unsigned u) {
  unsigned b = (u >> 31) ? (u ^ 0x80000000u) : ~u;
  return __uint_as_float(b);
}

// ---------------------------------------------------------------------------
// Weight packing: W[rows x cols] fp32 -> f16 B fragments [kChunks][nChunks].
// Fragment (c,n): lane holds column N = 16n + (lane&15), half h at
// K = 32c + kmap(h, lane>>4). permF1 applies the e_in K-permutation for Wf1:
// K 0..63 <- rows 3..66 (x part), K 64..66 <- rows 0..2 (rel part), else 0.
// ---------------------------------------------------------------------------
__global__ __launch_bounds__(32) void pack_w(const float* __restrict__ W,
                                             int rows, int cols, int nChunks,
                                             int permF1,
                                             _Float16* __restrict__ out) {
  const int lane = threadIdx.x;
  const int g = lane >> 4, r = lane & 15;
  const int c = blockIdx.x / nChunks;
  const int n = blockIdx.x % nChunks;
#pragma unroll
  for (int h = 0; h < 16; ++h) {
    int K = 32 * c + (h + 8 * g + (h >= 8 ? 8 : 0));
    int N = 16 * n + r;
    int sr;
    if (permF1) sr = (K < 64) ? (K + 3) : ((K < 67) ? (K - 64) : -1);
    else        sr = (K < rows) ? K : -1;
    float v = 0.f;
    if (sr >= 0 && N < cols) v = W[(size_t)sr * cols + N];
    out[((size_t)blockIdx.x * 32 + lane) * 16 + h] = (_Float16)v;
  }
}

__global__ void cvt_f16(const float* __restrict__ x, _Float16* __restrict__ xh,
                        int n) {
  for (int i = blockIdx.x * blockDim.x + threadIdx.x; i < n;
       i += gridDim.x * blockDim.x)
    xh[i] = (_Float16)x[i];
}

__global__ void fill_agg(unsigned* __restrict__ agg, int n) {
  unsigned v = encf(NEG_BIG);
  for (int i = blockIdx.x * blockDim.x + threadIdx.x; i < n;
       i += gridDim.x * blockDim.x)
    agg[i] = v;
}

// ---------------------------------------------------------------------------
// delta = mlp_h(x):  [16,64]@[64,64] ReLU @[64,3]  (Wh2 padded to 16 cols)
// ---------------------------------------------------------------------------
__global__ __launch_bounds__(32) void delta_kernel(
    const _Float16* __restrict__ xh,
    const _Float16* __restrict__ Wh1p, const float* __restrict__ bh1,
    const _Float16* __restrict__ Wh2p, const float* __restrict__ bh2,
    float* __restrict__ delta) {
  __shared__ __align__(16) _Float16 sHid[16][64];
  const int lane = threadIdx.x;
  const int g = lane >> 4, r = lane & 15;
  const int n0 = blockIdx.x * 16;

  v8f acc[4];
#pragma unroll
  for (int n = 0; n < 4; ++n) {
    float bv = bh1[n * 16 + r];
#pragma unroll
    for (int j = 0; j < 8; ++j) acc[n][j] = bv;
  }
  const _Float16* rowp = xh + (size_t)(n0 + r) * DD;
#pragma unroll
  for (int c = 0; c < 2; ++c) {
    v16h a = load_a16(rowp, c, g);
#pragma unroll
    for (int n = 0; n < 4; ++n) {
      v16h b = load_b16(Wh1p + ((size_t)(c * 4 + n) * 32 + lane) * 16);
      acc[n] = __builtin_amdgcn_wmma_f32_16x16x32_f16(false, a, false, b,
                                                      (short)0, acc[n], false,
                                                      false);
    }
  }
#pragma unroll
  for (int n = 0; n < 4; ++n)
#pragma unroll
    for (int j = 0; j < 8; ++j) {
      float v = acc[n][j];
      v = v > 0.f ? v : 0.f;
      sHid[j + 8 * g][n * 16 + r] = (_Float16)v;
    }
  wave_lds_sync();

  v8f o;
  {
    float bv = (r < 3) ? bh2[r] : 0.f;
#pragma unroll
    for (int j = 0; j < 8; ++j) o[j] = bv;
  }
#pragma unroll
  for (int c = 0; c < 2; ++c) {
    v16h a = load_a16(&sHid[r][0], c, g);
    v16h b = load_b16(Wh2p + ((size_t)c * 32 + lane) * 16);
    o = __builtin_amdgcn_wmma_f32_16x16x32_f16(false, a, false, b, (short)0, o,
                                               false, false);
  }
  if (r < 3) {
#pragma unroll
    for (int j = 0; j < 8; ++j)
      delta[(size_t)(n0 + j + 8 * g) * 3 + r] = o[j];
  }
}

// ---------------------------------------------------------------------------
// Edge kernel: e_in = [x_src(64) | rel(3) | pad(29)] (K-permuted),
// e = mlp_f(e_in) via WMMA, then fused segment_max via atomicMax(u32).
// One wave processes EDGE_TILES_PER_WAVE tiles of 16 edges; all 20 weight
// B-fragments stay VGPR-resident for the wave's lifetime, so the steady-state
// loop issues only gathers, ds loads/stores, WMMAs, and atomics.
// ---------------------------------------------------------------------------
__global__ __launch_bounds__(32) void edge_kernel(
    const _Float16* __restrict__ xh, const float* __restrict__ pos,
    const float* __restrict__ delta, const int* __restrict__ src,
    const int* __restrict__ dst,
    const _Float16* __restrict__ Wf1p, const float* __restrict__ bf1,
    const _Float16* __restrict__ Wf2p, const float* __restrict__ bf2,
    unsigned* __restrict__ aggU) {
  __shared__ __align__(16) _Float16 sEin[16][96];
  __shared__ __align__(16) _Float16 sHid[16][64];
  __shared__ int sDst[16];
  const int lane = threadIdx.x;
  const int g = lane >> 4, r = lane & 15;

  // Hoist weights into registers: 12 + 8 fragments (160 VGPRs).
  v16h B1[12];
#pragma unroll
  for (int i = 0; i < 12; ++i)
    B1[i] = load_b16(Wf1p + ((size_t)i * 32 + lane) * 16);
  v16h B2[8];
#pragma unroll
  for (int i = 0; i < 8; ++i)
    B2[i] = load_b16(Wf2p + ((size_t)i * 32 + lane) * 16);

  float b1v[4], b2v[4];
#pragma unroll
  for (int n = 0; n < 4; ++n) {
    b1v[n] = bf1[n * 16 + r];
    b2v[n] = bf2[n * 16 + r];
  }

  // Zero pad region of e_in once; it is reused by every tile.
  if (lane < 16)
    for (int k = 67; k < 96; ++k) sEin[lane][k] = (_Float16)0.f;

  const int tile0 = blockIdx.x * EDGE_TILES_PER_WAVE;
#pragma unroll 1
  for (int t = 0; t < EDGE_TILES_PER_WAVE; ++t) {
    const int e0 = (tile0 + t) * 16;

    // Stage e_in: lanes 0..15 compute rel + dst index; lanes 16..31 copy x[src].
    if (lane < 16) {
      int e = e0 + lane;
      int s = src[e], d = dst[e];
      sDst[lane] = d;
#pragma unroll
      for (int k = 0; k < 3; ++k) {
        float rel = pos[(size_t)s * 3 + k] - pos[(size_t)d * 3 + k] +
                    delta[(size_t)d * 3 + k];
        sEin[lane][64 + k] = (_Float16)rel;
      }
    } else {
      int rr = lane - 16;
      int s = src[e0 + rr];
      const v4f* px = (const v4f*)(xh + (size_t)s * DD);
      v4f* pl = (v4f*)(&sEin[rr][0]);
#pragma unroll
      for (int i = 0; i < 8; ++i) pl[i] = px[i];
    }
    wave_lds_sync();

    // GEMM1: [16,96] @ Wf1p[96,64]
    v8f acc[4];
#pragma unroll
    for (int n = 0; n < 4; ++n) {
#pragma unroll
      for (int j = 0; j < 8; ++j) acc[n][j] = b1v[n];
    }
#pragma unroll
    for (int c = 0; c < 3; ++c) {
      v16h a = load_a16(&sEin[r][0], c, g);
#pragma unroll
      for (int n = 0; n < 4; ++n)
        acc[n] = __builtin_amdgcn_wmma_f32_16x16x32_f16(
            false, a, false, B1[c * 4 + n], (short)0, acc[n], false, false);
    }
#pragma unroll
    for (int n = 0; n < 4; ++n)
#pragma unroll
      for (int j = 0; j < 8; ++j) {
        float v = acc[n][j];
        v = v > 0.f ? v : 0.f;
        sHid[j + 8 * g][n * 16 + r] = (_Float16)v;
      }
    wave_lds_sync();

    // GEMM2: [16,64] @ Wf2p[64,64]
    v8f o[4];
#pragma unroll
    for (int n = 0; n < 4; ++n) {
#pragma unroll
      for (int j = 0; j < 8; ++j) o[n][j] = b2v[n];
    }
#pragma unroll
    for (int c = 0; c < 2; ++c) {
      v16h a = load_a16(&sHid[r][0], c, g);
#pragma unroll
      for (int n = 0; n < 4; ++n)
        o[n] = __builtin_amdgcn_wmma_f32_16x16x32_f16(
            false, a, false, B2[c * 4 + n], (short)0, o[n], false, false);
    }

    // Fused segment_max: element (edge = j+8g, col = 16n+r)
#pragma unroll
    for (int j = 0; j < 8; ++j) {
      int d = sDst[j + 8 * g];
      unsigned* bp = aggU + (size_t)d * DD;
#pragma unroll
      for (int n = 0; n < 4; ++n)
        atomicMax(&bp[n * 16 + r], encf(o[n][j]));
    }
  }
}

// ---------------------------------------------------------------------------
// Node kernel: decode agg (empty -> 0), mlp_g via WMMA, residual add.
// Also emits the f16 copy of x for the next layer.
// ---------------------------------------------------------------------------
__global__ __launch_bounds__(32) void node_kernel(
    const unsigned* __restrict__ aggU, const float* __restrict__ xin,
    const _Float16* __restrict__ Wg1p, const float* __restrict__ bg1,
    const _Float16* __restrict__ Wg2p, const float* __restrict__ bg2,
    float* __restrict__ xout, _Float16* __restrict__ xhout) {
  __shared__ __align__(16) _Float16 sA[16][64];
  __shared__ __align__(16) _Float16 sHid[16][64];
  const int lane = threadIdx.x;
  const int g = lane >> 4, r = lane & 15;
  const int n0 = blockIdx.x * 16;

  for (int idx = lane; idx < 16 * 64; idx += 32) {
    int row = idx >> 6, col = idx & 63;
    float f = decf(aggU[(size_t)(n0 + row) * DD + col]);
    if (f <= NEG_BIG) f = 0.f;
    sA[row][col] = (_Float16)f;
  }
  wave_lds_sync();

  v8f acc[4];
#pragma unroll
  for (int n = 0; n < 4; ++n) {
    float bv = bg1[n * 16 + r];
#pragma unroll
    for (int j = 0; j < 8; ++j) acc[n][j] = bv;
  }
#pragma unroll
  for (int c = 0; c < 2; ++c) {
    v16h a = load_a16(&sA[r][0], c, g);
#pragma unroll
    for (int n = 0; n < 4; ++n) {
      v16h b = load_b16(Wg1p + ((size_t)(c * 4 + n) * 32 + lane) * 16);
      acc[n] = __builtin_amdgcn_wmma_f32_16x16x32_f16(false, a, false, b,
                                                      (short)0, acc[n], false,
                                                      false);
    }
  }
#pragma unroll
  for (int n = 0; n < 4; ++n)
#pragma unroll
    for (int j = 0; j < 8; ++j) {
      float v = acc[n][j];
      v = v > 0.f ? v : 0.f;
      sHid[j + 8 * g][n * 16 + r] = (_Float16)v;
    }
  wave_lds_sync();

  v8f o[4];
#pragma unroll
  for (int n = 0; n < 4; ++n) {
    float bv = bg2[n * 16 + r];
#pragma unroll
    for (int j = 0; j < 8; ++j) o[n][j] = bv;
  }
#pragma unroll
  for (int c = 0; c < 2; ++c) {
    v16h a = load_a16(&sHid[r][0], c, g);
#pragma unroll
    for (int n = 0; n < 4; ++n) {
      v16h b = load_b16(Wg2p + ((size_t)(c * 4 + n) * 32 + lane) * 16);
      o[n] = __builtin_amdgcn_wmma_f32_16x16x32_f16(false, a, false, b,
                                                    (short)0, o[n], false,
                                                    false);
    }
  }

#pragma unroll
  for (int n = 0; n < 4; ++n)
#pragma unroll
    for (int j = 0; j < 8; ++j) {
      size_t node = (size_t)(n0 + j + 8 * g);
      int col = n * 16 + r;
      float v = o[n][j] + xin[node * DD + col];
      xout[node * DD + col] = v;
      xhout[node * DD + col] = (_Float16)v;
    }
}

// ---------------------------------------------------------------------------
extern "C" void kernel_launch(void* const* d_in, const int* in_sizes, int n_in,
                              void* d_out, int out_size, void* d_ws,
                              size_t ws_size, hipStream_t stream) {
  (void)out_size; (void)ws_size;
  // Locate x / pos / edge_index by unique flat sizes; collect params in order.
  const float* x = nullptr;
  const float* pos = nullptr;
  const int* ei = nullptr;
  int pidx[36];
  int np = 0;
  for (int i = 0; i < n_in; ++i) {
    int s = in_sizes[i];
    if (s == N_NODES * DD && x == nullptr)       x = (const float*)d_in[i];
    else if (s == N_NODES * 3 && pos == nullptr) pos = (const float*)d_in[i];
    else if (s == 2 * N_EDGES && ei == nullptr)  ei = (const int*)d_in[i];
    else if (np < 36)                            pidx[np++] = i;
  }
  // Per-layer params in insertion order:
  // 0:Wh1 1:bh1 2:Wh2 3:bh2 4:Wf1 5:bf1 6:Wf2 7:bf2 8:Wg1 9:bg1 10:Wg2 11:bg2
  // If the harness flattened via jax pytrees, dict keys are sorted:
  // [Wf1,Wf2,Wg1,Wg2,Wh1,Wh2,bf1,bf2,bg1,bg2,bh1,bh2] -> detect via Wf1=4288.
  const float* P[3][12];
  bool sorted_layout = (np == 36) && (in_sizes[pidx[0]] == 4288);
  static const int smap[12] = {4, 10, 5, 11, 0, 6, 1, 7, 2, 8, 3, 9};
  for (int l = 0; l < 3; ++l)
    for (int k = 0; k < 12; ++k) {
      int j = sorted_layout ? smap[k] : k;
      P[l][k] = (const float*)d_in[pidx[l * 12 + j]];
    }

  // Workspace layout (~52 MB total)
  char* ws = (char*)d_ws;
  size_t off = 0;
  auto take = [&](size_t bytes) {
    size_t r0 = off;
    off = (off + bytes + 255) & ~(size_t)255;
    return r0;
  };
  size_t packOff = take((size_t)3 * 46 * 1024);       // packed f16 weights
  size_t deltaOff = take((size_t)N_NODES * 3 * 4);    // delta [N,3] f32
  size_t aggOff = take((size_t)N_NODES * DD * 4);     // encoded max [N,64] u32
  size_t xhAOff = take((size_t)N_NODES * DD * 2);     // x f16 ping
  size_t xhBOff = take((size_t)N_NODES * DD * 2);     // x f16 pong
  size_t x1Off = take((size_t)N_NODES * DD * 4);      // x after layer 0
  size_t x2Off = take((size_t)N_NODES * DD * 4);      // x after layer 1

  // Pack all weights (once per call; 18 tiny launches)
  for (int l = 0; l < 3; ++l) {
    _Float16* packL = (_Float16*)(ws + packOff) + (size_t)l * 46 * 512;
    _Float16* Wh1p = packL;            // 8 frags
    _Float16* Wh2p = packL + 8 * 512;  // 2 frags
    _Float16* Wf1p = packL + 10 * 512; // 12 frags
    _Float16* Wf2p = packL + 22 * 512; // 8 frags
    _Float16* Wg1p = packL + 30 * 512; // 8 frags
    _Float16* Wg2p = packL + 38 * 512; // 8 frags
    pack_w<<<8, 32, 0, stream>>>(P[l][0], 64, 64, 4, 0, Wh1p);
    pack_w<<<2, 32, 0, stream>>>(P[l][2], 64, 3, 1, 0, Wh2p);
    pack_w<<<12, 32, 0, stream>>>(P[l][4], 67, 64, 4, 1, Wf1p);
    pack_w<<<8, 32, 0, stream>>>(P[l][6], 64, 64, 4, 0, Wf2p);
    pack_w<<<8, 32, 0, stream>>>(P[l][8], 64, 64, 4, 0, Wg1p);
    pack_w<<<8, 32, 0, stream>>>(P[l][10], 64, 64, 4, 0, Wg2p);
  }

  cvt_f16<<<4096, 256, 0, stream>>>(x, (_Float16*)(ws + xhAOff), N_NODES * DD);

  const int* srcI = ei;
  const int* dstI = ei + N_EDGES;
  float* deltaP = (float*)(ws + deltaOff);
  unsigned* aggU = (unsigned*)(ws + aggOff);

  for (int l = 0; l < 3; ++l) {
    _Float16* packL = (_Float16*)(ws + packOff) + (size_t)l * 46 * 512;
    _Float16* Wh1p = packL;
    _Float16* Wh2p = packL + 8 * 512;
    _Float16* Wf1p = packL + 10 * 512;
    _Float16* Wf2p = packL + 22 * 512;
    _Float16* Wg1p = packL + 30 * 512;
    _Float16* Wg2p = packL + 38 * 512;

    const _Float16* xh_in = (const _Float16*)(ws + ((l & 1) ? xhBOff : xhAOff));
    _Float16* xh_out = (_Float16*)(ws + ((l & 1) ? xhAOff : xhBOff));
    const float* xin =
        (l == 0) ? x : (const float*)(ws + (l == 1 ? x1Off : x2Off));
    float* xout =
        (l == 2) ? (float*)d_out : (float*)(ws + (l == 0 ? x1Off : x2Off));

    delta_kernel<<<N_NODES / 16, 32, 0, stream>>>(xh_in, Wh1p, P[l][1], Wh2p,
                                                  P[l][3], deltaP);
    fill_agg<<<(N_NODES * DD + 255) / 256, 256, 0, stream>>>(aggU,
                                                             N_NODES * DD);
    edge_kernel<<<N_EDGES / 16 / EDGE_TILES_PER_WAVE, 32, 0, stream>>>(
        xh_in, pos, deltaP, srcI, dstI, Wf1p, P[l][5], Wf2p, P[l][7], aggU);
    node_kernel<<<N_NODES / 16, 32, 0, stream>>>(aggU, xin, Wg1p, P[l][9],
                                                 Wg2p, P[l][11], xout, xh_out);
  }
}
